// Model_43636867727542
// MI455X (gfx1250) — matrix operand order repaired and checked by reading, not compiled, not run on hardware
//
#include <hip/hip_runtime.h>

// ---------------------------------------------------------------------------
// CDNA5 (gfx1250, wave32) hetero-GraphSAGE.
// Dense math: v_wmma_f32_16x16x32_bf16 (fp32 accum).  Irregular math:
// fp32 global atomics for segment-mean.  Decoder fully fused in-register.
// GEMM: 32Mx64N per wave (B fragments shared by two A fragments).
// ---------------------------------------------------------------------------

typedef __attribute__((ext_vector_type(16))) __bf16 v16bf;
typedef __attribute__((ext_vector_type(8)))  float  v8f;

static const int CN_NEWS = 50000;
static const int CN_KW   = 10000;
static const int CN_ST   = 2000;
static const int CF_NEWS = 768;
static const int CF_KW   = 128;
static const int CF_ST   = 64;
static const int CHID    = 256;
static const int COUT    = 128;
static const int CE_HK   = 500000;
static const int CE_HS   = 250000;
static const int CE_LBL  = 150000;

#define FLAG_ACC  1
#define FLAG_RELU 2

// ---------------------------------------------------------------------------
// f32 -> bf16 conversion (one-time weight repack)
// ---------------------------------------------------------------------------
__global__ void cvt_f32_to_bf16(const float* __restrict__ s, __bf16* __restrict__ d, int n) {
  int i = blockIdx.x * blockDim.x + threadIdx.x;
  if (i < n) d[i] = (__bf16)s[i];
}

// ---------------------------------------------------------------------------
// Segment mean: scatter-add + count, then divide.  F fixed at CHID=256.
// ---------------------------------------------------------------------------
__global__ void scatter_add_feat(const float* __restrict__ x,
                                 const int* __restrict__ src,
                                 const int* __restrict__ dst,
                                 float* __restrict__ m,
                                 float* __restrict__ cnt) {
  int e = blockIdx.x;
  int f = threadIdx.x;
  int s = src[e];
  int d = dst[e];
  __hip_atomic_fetch_add(&m[(size_t)d * CHID + f], x[(size_t)s * CHID + f],
                         __ATOMIC_RELAXED, __HIP_MEMORY_SCOPE_AGENT);
  if (f == 0)
    __hip_atomic_fetch_add(&cnt[d], 1.0f, __ATOMIC_RELAXED, __HIP_MEMORY_SCOPE_AGENT);
}

__global__ void seg_divide(float* __restrict__ m, const float* __restrict__ cnt,
                           long total) {
  long i = (long)blockIdx.x * blockDim.x + threadIdx.x;
  if (i < total) {
    float c = cnt[i >> 8];   // F == 256
    m[i] = m[i] / fmaxf(c, 1.0f);
  }
}

// ---------------------------------------------------------------------------
// WMMA fragment helpers.
//
// A (16x32 bf16, row-major fp32 source, ld=K):
//   lane<16  : row = lane,    K = k0..k0+15 (contiguous)
//   lane>=16 : row = lane-16, K = k0+16..k0+31 (contiguous)
// B (32x16 bf16, from W[N][K] row-major bf16):
//   lane L holds 16 contiguous K values of column n0+(L&15),
//   starting at k0 (L<16) or k0+16 (L>=16).
// C/D f32 16x16: VGPR r, lane L -> row r+((L>>4)<<3), col L&15.
// ---------------------------------------------------------------------------
__device__ __forceinline__ void cvt8(const float* __restrict__ p, v16bf& a, int base) {
#pragma unroll
  for (int i = 0; i < 8; ++i) a[base + i] = (__bf16)p[i];
}

// ---------------------------------------------------------------------------
// out[M,N] = act( A[M,K]_f32 @ Wbf16[N,K]^T + bias[N] (+ out prev) )
// one wave per 32(M) x 64(N) tile.  grid = (ceil(M/32), N/64), block = 32.
// A rows clamped for the ragged last tile; stores guarded (post-WMMA only).
// ---------------------------------------------------------------------------
__global__ void __launch_bounds__(32)
gemm_bf16_wmma(const float* __restrict__ A, const __bf16* __restrict__ W,
               const float* __restrict__ bias, float* __restrict__ out,
               int M, int N, int K, int flags) {
  const int lane = threadIdx.x;
  const int m0 = blockIdx.x << 5;
  const int n0 = blockIdx.y << 6;

  v8f c[8] = {};   // c[t] for rows m0..m0+15, c[4+t] for rows m0+16..m0+31

  int r0 = m0 + (lane & 15);
  int r1 = r0 + 16;
  if (r0 > M - 1) r0 = M - 1;
  if (r1 > M - 1) r1 = M - 1;
  const int koff = (lane < 16) ? 0 : 16;
  const float* arow0 = A + (size_t)r0 * K + koff;
  const float* arow1 = A + (size_t)r1 * K + koff;
  const __bf16* wp = W + (size_t)(n0 + (lane & 15)) * K + ((lane >> 4) << 4);

  for (int k0 = 0; k0 < K; k0 += 32) {
    v16bf a0, a1;
    cvt8(arow0 + k0, a0, 0);
    cvt8(arow0 + k0 + 8, a0, 8);
    cvt8(arow1 + k0, a1, 0);
    cvt8(arow1 + k0 + 8, a1, 8);
    if (k0 + 32 < K) {
      __builtin_prefetch(arow0 + k0 + 32, 0, 1);
      __builtin_prefetch(arow1 + k0 + 32, 0, 1);
    }
#pragma unroll
    for (int t = 0; t < 4; ++t) {
      v16bf b = *(const v16bf*)(wp + (size_t)t * 16 * K + k0);
      c[t]     = __builtin_amdgcn_wmma_f32_16x16x32_bf16(false, a0, false, b, (short)0, c[t],     false, false);
      c[4 + t] = __builtin_amdgcn_wmma_f32_16x16x32_bf16(false, a1, false, b, (short)0, c[4 + t], false, false);
    }
  }

#pragma unroll
  for (int h = 0; h < 2; ++h) {
    const int rbase = m0 + (h << 4) + ((lane >> 4) << 3);
#pragma unroll
    for (int t = 0; t < 4; ++t) {
      const int cn = n0 + (t << 4) + (lane & 15);
      const float bv = bias ? bias[cn] : 0.0f;
      float* op = out + (size_t)rbase * N + cn;
#pragma unroll
      for (int r = 0; r < 8; ++r) {
        if (rbase + r < M) {
          float v = c[(h << 2) + t][r] + bv;
          if (flags & FLAG_ACC) v += op[(size_t)r * N];
          if (flags & FLAG_RELU) v = fmaxf(v, 0.0f);
          op[(size_t)r * N] = v;
        }
      }
    }
  }
}

// ---------------------------------------------------------------------------
// Fused edge decoder: 16 label edges / wave.
// A row = [nodeA[src] (128) | nodeB[dst] (128)]  (K=256), loaded ONCE into
// 8 register-resident bf16 fragments, then 8 N-tiles of WMMA + relu + dot(w2).
// ---------------------------------------------------------------------------
__global__ void __launch_bounds__(32)
decoder_fused(const float* __restrict__ na, const float* __restrict__ nb,
              const int* __restrict__ src, const int* __restrict__ dst,
              const __bf16* __restrict__ W1, const float* __restrict__ b1,
              const float* __restrict__ w2, const float* __restrict__ b2,
              float* __restrict__ out) {
  const int lane = threadIdx.x;
  const int e0 = blockIdx.x << 4;
  const int row = lane & 15;

  const int si = src[e0 + row];
  const int di = dst[e0 + row];
  const float* ra = na + (size_t)si * COUT;   // first 128 features
  const float* rb = nb + (size_t)di * COUT;   // second 128 features

  const int koff = (lane < 16) ? 0 : 16;
  const __bf16* wp = W1 + (size_t)(lane & 15) * 256 + ((lane >> 4) << 4);
  const float bb = b2[0];

  // gather the full K=256 A row into 8 bf16 fragments (one global read of A)
  v16bf av[8];
#pragma unroll
  for (int kt = 0; kt < 8; ++kt) {
    const int k0 = kt * 32;
    const int ka = k0 + koff;       // chunk 0 start (8 elems)
    const int kb = ka + 8;          // chunk 1 start (8 elems)
    const float* pa = (ka < COUT) ? (ra + ka) : (rb + (ka - COUT));
    const float* pb = (kb < COUT) ? (ra + kb) : (rb + (kb - COUT));
    cvt8(pa, av[kt], 0);
    cvt8(pb, av[kt], 8);
  }

  float partial[8] = {0.f, 0.f, 0.f, 0.f, 0.f, 0.f, 0.f, 0.f};

  for (int nt = 0; nt < 8; ++nt) {
    v8f c = {};
#pragma unroll
    for (int kt = 0; kt < 8; ++kt) {
      v16bf b = *(const v16bf*)(wp + (size_t)nt * 16 * 256 + kt * 32);
      c = __builtin_amdgcn_wmma_f32_16x16x32_bf16(false, av[kt], false, b, (short)0, c, false, false);
    }
    const int cn = (nt << 4) + (lane & 15);
    const float bv = b1[cn];
    const float wv = w2[cn];
#pragma unroll
    for (int r = 0; r < 8; ++r) {
      float h = c[r] + bv;
      h = fmaxf(h, 0.0f);
      partial[r] += h * wv;
    }
  }

  // reduce the 16 columns held within each half-wave
#pragma unroll
  for (int r = 0; r < 8; ++r) {
    float v = partial[r];
    v += __shfl_xor(v, 1);
    v += __shfl_xor(v, 2);
    v += __shfl_xor(v, 4);
    v += __shfl_xor(v, 8);
    partial[r] = v;
  }
  if ((lane & 15) == 0) {
    const int rb2 = (lane >> 4) << 3;
#pragma unroll
    for (int r = 0; r < 8; ++r) out[e0 + rb2 + r] = partial[r] + bb;
  }
}

// ---------------------------------------------------------------------------
// Host orchestration
// ---------------------------------------------------------------------------
extern "C" void kernel_launch(void* const* d_in, const int* in_sizes, int n_in,
                              void* d_out, int out_size, void* d_ws, size_t ws_size,
                              hipStream_t stream) {
  (void)in_sizes; (void)n_in; (void)out_size; (void)ws_size;

  const float* x_news = (const float*)d_in[0];
  const float* x_kw   = (const float*)d_in[1];
  const float* x_st   = (const float*)d_in[2];
  const float* Wn = (const float*)d_in[3];  const float* bn = (const float*)d_in[4];
  const float* Wk = (const float*)d_in[5];  const float* bk = (const float*)d_in[6];
  const float* Ws = (const float*)d_in[7];  const float* bs = (const float*)d_in[8];
  const float* W1l = (const float*)d_in[9];
  const float* b1  = (const float*)d_in[10];
  const float* W1r = (const float*)d_in[11];
  const float* W2l = (const float*)d_in[12];
  const float* b2  = (const float*)d_in[13];
  const float* W2r = (const float*)d_in[14];
  const float* DW1 = (const float*)d_in[15];
  const float* Db1 = (const float*)d_in[16];
  const float* DW2 = (const float*)d_in[17];
  const float* Db2 = (const float*)d_in[18];
  const int* e_hk_src = (const int*)d_in[19];
  const int* e_hk_dst = (const int*)d_in[20];
  const int* e_hs_src = (const int*)d_in[21];
  const int* e_hs_dst = (const int*)d_in[22];
  const int* lbl_hk_src = (const int*)d_in[23];
  const int* lbl_hk_dst = (const int*)d_in[24];
  const int* lbl_hs_src = (const int*)d_in[25];
  const int* lbl_hs_dst = (const int*)d_in[26];

  float* out = (float*)d_out;
  char* ws = (char*)d_ws;

  // ---- workspace carve (256B aligned) ----
  size_t off = 0;
  auto alloc = [&](size_t bytes) -> char* {
    char* p = ws + off;
    off = (off + bytes + 255) & ~(size_t)255;
    return p;
  };
  __bf16* bWn  = (__bf16*)alloc((size_t)CHID * CF_NEWS * 2);
  __bf16* bWk  = (__bf16*)alloc((size_t)CHID * CF_KW * 2);
  __bf16* bWs  = (__bf16*)alloc((size_t)CHID * CF_ST * 2);
  __bf16* bW1l = (__bf16*)alloc((size_t)4 * CHID * CHID * 2);
  __bf16* bW1r = (__bf16*)alloc((size_t)4 * CHID * CHID * 2);
  __bf16* bW2l = (__bf16*)alloc((size_t)4 * COUT * CHID * 2);
  __bf16* bW2r = (__bf16*)alloc((size_t)4 * COUT * CHID * 2);
  __bf16* bDW1 = (__bf16*)alloc((size_t)2 * COUT * (2 * COUT) * 2);

  float* hn = (float*)alloc((size_t)CN_NEWS * CHID * 4);
  float* hk = (float*)alloc((size_t)CN_KW * CHID * 4);
  float* hs = (float*)alloc((size_t)CN_ST * CHID * 4);
  float* k1 = (float*)alloc((size_t)CN_KW * CHID * 4);
  float* n1 = (float*)alloc((size_t)CN_NEWS * CHID * 4);
  float* s1 = (float*)alloc((size_t)CN_ST * CHID * 4);
  float* m   = (float*)alloc((size_t)CN_NEWS * CHID * 4);
  float* cnt = (float*)alloc((size_t)CN_NEWS * 4);
  // layer-2 outputs alias the (dead after layer-1) projection buffers
  float* n2 = hn;   // 50000 x 128  <= 50000 x 256
  float* k2 = hk;   // 10000 x 128
  float* s2 = hs;   //  2000 x 128

  // ---- weight repack to bf16 ----
  auto cvt = [&](const float* s, __bf16* d, int n) {
    cvt_f32_to_bf16<<<(n + 255) / 256, 256, 0, stream>>>(s, d, n);
  };
  cvt(Wn, bWn, CHID * CF_NEWS);
  cvt(Wk, bWk, CHID * CF_KW);
  cvt(Ws, bWs, CHID * CF_ST);
  cvt(W1l, bW1l, 4 * CHID * CHID);
  cvt(W1r, bW1r, 4 * CHID * CHID);
  cvt(W2l, bW2l, 4 * COUT * CHID);
  cvt(W2r, bW2r, 4 * COUT * CHID);
  cvt(DW1, bDW1, 2 * COUT * 2 * COUT);

  auto gemm = [&](const float* A, const __bf16* W, const float* bias,
                  float* o, int M, int N, int K, int flags) {
    dim3 grid((M + 31) / 32, N / 64);
    gemm_bf16_wmma<<<grid, 32, 0, stream>>>(A, W, bias, o, M, N, K, flags);
  };

  auto agg = [&](const float* xs, const int* s, const int* d, int E, int nd) {
    hipMemsetAsync(m, 0, (size_t)nd * CHID * sizeof(float), stream);
    hipMemsetAsync(cnt, 0, (size_t)nd * sizeof(float), stream);
    scatter_add_feat<<<E, CHID, 0, stream>>>(xs, s, d, m, cnt);
    long total = (long)nd * CHID;
    seg_divide<<<(unsigned)((total + 255) / 256), 256, 0, stream>>>(m, cnt, total);
  };

  // ---- input projections ----
  gemm(x_news, bWn, bn, hn, CN_NEWS, CHID, CF_NEWS, FLAG_RELU);
  gemm(x_kw,   bWk, bk, hk, CN_KW,   CHID, CF_KW,   FLAG_RELU);
  gemm(x_st,   bWs, bs, hs, CN_ST,   CHID, CF_ST,   FLAG_RELU);

  const size_t s1w = (size_t)CHID * CHID;  // per-relation stride, layer 1
  const size_t s2w = (size_t)COUT * CHID;  // per-relation stride, layer 2

  // ---- layer 1 (relu at end of each dst-type accumulation) ----
  // k1 = relu(mean(hn over hk) @ W1l0 + b1[0] + hk @ W1r0)
  agg(hn, e_hk_src, e_hk_dst, CE_HK, CN_KW);
  gemm(m,  bW1l + 0 * s1w, b1 + 0 * CHID, k1, CN_KW, CHID, CHID, 0);
  gemm(hk, bW1r + 0 * s1w, nullptr,       k1, CN_KW, CHID, CHID, FLAG_ACC | FLAG_RELU);
  // n1 = relu(rev-hk SAGE + rev-hs SAGE)
  agg(hk, e_hk_dst, e_hk_src, CE_HK, CN_NEWS);
  gemm(m,  bW1l + 1 * s1w, b1 + 1 * CHID, n1, CN_NEWS, CHID, CHID, 0);
  gemm(hn, bW1r + 1 * s1w, nullptr,       n1, CN_NEWS, CHID, CHID, FLAG_ACC);
  agg(hs, e_hs_dst, e_hs_src, CE_HS, CN_NEWS);
  gemm(m,  bW1l + 3 * s1w, b1 + 3 * CHID, n1, CN_NEWS, CHID, CHID, FLAG_ACC);
  gemm(hn, bW1r + 3 * s1w, nullptr,       n1, CN_NEWS, CHID, CHID, FLAG_ACC | FLAG_RELU);
  // s1
  agg(hn, e_hs_src, e_hs_dst, CE_HS, CN_ST);
  gemm(m,  bW1l + 2 * s1w, b1 + 2 * CHID, s1, CN_ST, CHID, CHID, 0);
  gemm(hs, bW1r + 2 * s1w, nullptr,       s1, CN_ST, CHID, CHID, FLAG_ACC | FLAG_RELU);

  // ---- layer 2 (no relu) ----
  agg(n1, e_hk_src, e_hk_dst, CE_HK, CN_KW);
  gemm(m,  bW2l + 0 * s2w, b2 + 0 * COUT, k2, CN_KW, COUT, CHID, 0);
  gemm(k1, bW2r + 0 * s2w, nullptr,       k2, CN_KW, COUT, CHID, FLAG_ACC);

  agg(k1, e_hk_dst, e_hk_src, CE_HK, CN_NEWS);
  gemm(m,  bW2l + 1 * s2w, b2 + 1 * COUT, n2, CN_NEWS, COUT, CHID, 0);
  gemm(n1, bW2r + 1 * s2w, nullptr,       n2, CN_NEWS, COUT, CHID, FLAG_ACC);
  agg(s1, e_hs_dst, e_hs_src, CE_HS, CN_NEWS);
  gemm(m,  bW2l + 3 * s2w, b2 + 3 * COUT, n2, CN_NEWS, COUT, CHID, FLAG_ACC);
  gemm(n1, bW2r + 3 * s2w, nullptr,       n2, CN_NEWS, COUT, CHID, FLAG_ACC);

  agg(n1, e_hs_src, e_hs_dst, CE_HS, CN_ST);
  gemm(m,  bW2l + 2 * s2w, b2 + 2 * COUT, s2, CN_ST, COUT, CHID, 0);
  gemm(s1, bW2r + 2 * s2w, nullptr,       s2, CN_ST, COUT, CHID, FLAG_ACC);

  // ---- fused decoders ----
  decoder_fused<<<CE_LBL / 16, 32, 0, stream>>>(
      n2, k2, lbl_hk_src, lbl_hk_dst,
      bDW1 + 0 * (size_t)COUT * 2 * COUT, Db1 + 0 * COUT, DW2 + 0 * COUT, Db2 + 0,
      out + 0);
  decoder_fused<<<CE_LBL / 16, 32, 0, stream>>>(
      n2, s2, lbl_hs_src, lbl_hs_dst,
      bDW1 + 1 * (size_t)COUT * 2 * COUT, Db1 + 1 * COUT, DW2 + 1 * COUT, Db2 + 1,
      out + CE_LBL);
}
